// GateLoopBlock_15290083574182
// MI455X (gfx1250) — matrix operand order repaired and checked by reading, not compiled, not run on hardware
//
#include <hip/hip_runtime.h>
#include <hip/hip_bf16.h>

// ---------------------------------------------------------------------------
// GateLoop block for MI455X (gfx1250, wave32, WMMA bf16)
// ---------------------------------------------------------------------------

#define Bb 2
#define Tt 4096
#define Dd 1024
#define Ff 4096
#define Mrows (Bb * Tt)          // 8192
#define SEG 32                   // scan segments per sequence
#define SEGLEN (Tt / SEG)        // 128

typedef __attribute__((ext_vector_type(16))) __bf16 v16bf;
typedef __attribute__((ext_vector_type(8)))  float  v8f;

union FragAB { v16bf v; uint4 q[2]; };

enum { EPI_STORE = 0, EPI_BIAS_GELU_BF16 = 1, EPI_BIAS_F32 = 2 };

static __device__ __forceinline__ unsigned short f2bf(float f) {
  union { __bf16 h; unsigned short u; } cv;
  cv.h = (__bf16)f;
  return cv.u;
}

static __device__ __forceinline__ float gelu_tanh(float x) {
  float x3 = x * x * x;
  float t = tanhf(0.7978845608028654f * (x + 0.044715f * x3));
  return 0.5f * x * (1.0f + t);
}

// a_complex = sigmoid(|a|) * exp(i*angle(a)) = (sigmoid(r)/r) * (ar + i*ai)
static __device__ __forceinline__ void ac_from(float ar, float ai,
                                               float& acr, float& aci) {
  float r = sqrtf(ar * ar + ai * ai);
  float sg = 1.0f / (1.0f + expf(-r));
  if (r > 1e-20f) {
    float sc = sg / r;
    acr = sc * ar;
    aci = sc * ai;
  } else {
    acr = 0.5f;   // sigmoid(0)*exp(i*0)
    aci = 0.0f;
  }
}

// ---------------------------------------------------------------------------
// f32 -> bf16 flat conversion (vectorized float4 -> 4x bf16)
// ---------------------------------------------------------------------------
__global__ __launch_bounds__(256) void k_convert_bf16(
    const float4* __restrict__ in, ushort4* __restrict__ out, int n4) {
  int i = blockIdx.x * 256 + threadIdx.x;
  if (i < n4) {
    float4 v = in[i];
    ushort4 o;
    o.x = f2bf(v.x); o.y = f2bf(v.y); o.z = f2bf(v.z); o.w = f2bf(v.w);
    out[i] = o;
  }
}

// ---------------------------------------------------------------------------
// Tiled transpose + convert + scale:
//   out[n * ldOut + kOff + k] = (bf16)(scale * in[k * N + n])
// ---------------------------------------------------------------------------
__global__ __launch_bounds__(256) void k_transpose_bf16(
    const float* __restrict__ in, __bf16* __restrict__ out, int K, int N,
    int ldOut, int kOff, float scale) {
  __shared__ float tile[32][33];
  int n0 = blockIdx.x * 32, k0 = blockIdx.y * 32;
  int tx = threadIdx.x & 31, ty = threadIdx.x >> 5;  // ty in 0..7
#pragma unroll
  for (int i = 0; i < 4; i++) {
    int k = k0 + ty + i * 8;
    tile[ty + i * 8][tx] = in[(size_t)k * N + n0 + tx];
  }
  __syncthreads();
#pragma unroll
  for (int i = 0; i < 4; i++) {
    int n = n0 + ty + i * 8;
    out[(size_t)n * ldOut + kOff + k0 + tx] =
        (__bf16)(scale * tile[tx][ty + i * 8]);
  }
}

// ---------------------------------------------------------------------------
// bf16 WMMA GEMM: C[M,N] = A[M,K] (row-major) * Bt[N,K]^T (Bt row-major = B^T)
// Block: 256 threads = 8 waves; block tile 256x128; wave tile 64x64; BK=32.
// Per K-step per wave: 16 ds_load_b128 -> 16 v_wmma (1.0 ds/wmma).
// ---------------------------------------------------------------------------
template <int EPI>
__global__ __launch_bounds__(256) void k_gemm_bf16_wmma(
    const __bf16* __restrict__ A, const __bf16* __restrict__ Bt,
    float* __restrict__ Cf, unsigned short* __restrict__ Cbf,
    const float* __restrict__ bias, int M, int N, int K) {
  constexpr int BM = 256, BN = 128, BK = 32, LDT = 40;  // padded lds row (bf16)
  __shared__ __align__(16) __bf16 As[BM * LDT];
  __shared__ __align__(16) __bf16 Bs[BN * LDT];

  const int tid  = threadIdx.x;
  const int lane = tid & 31;
  const int wid  = tid >> 5;
  const int wm   = wid >> 1;       // 0..3  (M direction, 64 rows each)
  const int wn   = wid & 1;        // 0..1  (N direction, 64 cols each)
  const int half = lane >> 4;      // 0..1
  const int lr   = lane & 15;      // 0..15

  const size_t m0 = (size_t)blockIdx.y * BM;
  const size_t n0 = (size_t)blockIdx.x * BN;

  v8f c[4][4];
#pragma unroll
  for (int mi = 0; mi < 4; mi++)
#pragma unroll
    for (int ni = 0; ni < 4; ni++) {
      v8f z = {0.f, 0.f, 0.f, 0.f, 0.f, 0.f, 0.f, 0.f};
      c[mi][ni] = z;
    }

  // Cooperative tile fill: each thread owns 16-element (32B) runs.
  const int fr = tid >> 1;              // 0..127 (tile row)
  const int fc = (tid & 1) * 16;        // 0 or 16 (k within tile)
  const __bf16* aP0 = A + (m0 + fr) * (size_t)K + fc;
  const __bf16* aP1 = aP0 + (size_t)128 * K;
  const __bf16* bP  = Bt + (n0 + fr) * (size_t)K + fc;

  for (int kk = 0; kk < K; kk += BK) {
    uint4 a00 = *(const uint4*)(aP0);
    uint4 a01 = *(const uint4*)(aP0 + 8);
    uint4 a10 = *(const uint4*)(aP1);
    uint4 a11 = *(const uint4*)(aP1 + 8);
    uint4 b0  = *(const uint4*)(bP);
    uint4 b1  = *(const uint4*)(bP + 8);
    // Prefetch two K-tiles ahead (gfx1250 global_prefetch_b8; speculative).
    __builtin_prefetch((const void*)(aP0 + 2 * BK), 0, 1);
    __builtin_prefetch((const void*)(aP1 + 2 * BK), 0, 1);
    __builtin_prefetch((const void*)(bP + 2 * BK), 0, 1);

    *(uint4*)&As[fr * LDT + fc]             = a00;
    *(uint4*)&As[fr * LDT + fc + 8]         = a01;
    *(uint4*)&As[(fr + 128) * LDT + fc]     = a10;
    *(uint4*)&As[(fr + 128) * LDT + fc + 8] = a11;
    *(uint4*)&Bs[fr * LDT + fc]             = b0;
    *(uint4*)&Bs[fr * LDT + fc + 8]         = b1;
    __syncthreads();

    // A fragments: lane half=0 -> K 0..7 & 16..23 ; half=1 -> K 8..15 & 24..31
    FragAB af[4];
#pragma unroll
    for (int mi = 0; mi < 4; mi++) {
      int row  = wm * 64 + mi * 16 + lr;
      int base = row * LDT + half * 8;
      af[mi].q[0] = *(const uint4*)&As[base];
      af[mi].q[1] = *(const uint4*)&As[base + 16];
    }
    // B fragments (Bt row = output column n): half=0 -> K 0..7 & 16..23 ; half=1 -> K 8..15 & 24..31
    FragAB bfr[4];
#pragma unroll
    for (int ni = 0; ni < 4; ni++) {
      int row  = wn * 64 + ni * 16 + lr;
      int base = row * LDT + half * 8;
      bfr[ni].q[0] = *(const uint4*)&Bs[base];
      bfr[ni].q[1] = *(const uint4*)&Bs[base + 16];
    }

#pragma unroll
    for (int mi = 0; mi < 4; mi++)
#pragma unroll
      for (int ni = 0; ni < 4; ni++)
        c[mi][ni] = __builtin_amdgcn_wmma_f32_16x16x32_bf16(
            false, af[mi].v, false, bfr[ni].v, (short)0, c[mi][ni],
            false, false);

    __syncthreads();
    aP0 += BK;
    aP1 += BK;
    bP  += BK;
  }

  // Epilogue: C/D layout — VGPR r holds M = r + 8*half, N = lane&15.
#pragma unroll
  for (int mi = 0; mi < 4; mi++)
#pragma unroll
    for (int ni = 0; ni < 4; ni++) {
      size_t n = n0 + (size_t)(wn * 64 + ni * 16 + lr);
#pragma unroll
      for (int r = 0; r < 8; r++) {
        size_t m = m0 + (size_t)(wm * 64 + mi * 16 + half * 8 + r);
        float val = c[mi][ni][r];
        size_t off = m * (size_t)N + n;
        if constexpr (EPI == EPI_STORE) {
          Cf[off] = val;
        } else if constexpr (EPI == EPI_BIAS_GELU_BF16) {
          Cbf[off] = f2bf(gelu_tanh(val + bias[n]));
        } else {  // EPI_BIAS_F32
          Cf[off] = val + bias[n];
        }
      }
    }
}

// ---------------------------------------------------------------------------
// GateLoop chunked scan — phase 1: per-segment carries (A-product, local h).
// qkvg[row][4d..4d+3] = (q,k,v,g);  a2[row][2d..2d+1] = (a_re, a_im)
// ---------------------------------------------------------------------------
__global__ __launch_bounds__(256) void k_scan_partial(
    const float4* __restrict__ qkvg, const float2* __restrict__ a2,
    float* __restrict__ cAr, float* __restrict__ cAi,
    float* __restrict__ cHr, float* __restrict__ cHi) {
  int tid  = blockIdx.x * 256 + threadIdx.x;   // B*D*SEG threads
  int d    = tid & (Dd - 1);
  int rest = tid >> 10;
  int b    = rest & (Bb - 1);
  int s    = rest >> 1;
  if (s >= SEG) return;

  float par = 1.f, pai = 0.f, hr = 0.f, hi = 0.f;
  size_t rowBase = (size_t)(b * Tt + s * SEGLEN);
  for (int t = 0; t < SEGLEN; t++) {
    size_t row = rowBase + t;
    float4 v4 = qkvg[row * Dd + d];
    float2 av = a2[row * Dd + d];
    float acr, aci;
    ac_from(av.x, av.y, acr, aci);
    float kv  = v4.y * v4.z;
    float nhr = acr * hr - aci * hi + kv;
    float nhi = acr * hi + aci * hr;
    hr = nhr; hi = nhi;
    float npr = acr * par - aci * pai;
    float npi = acr * pai + aci * par;
    par = npr; pai = npi;
  }
  int ch  = b * Dd + d;
  int idx = s * (Bb * Dd) + ch;
  cAr[idx] = par; cAi[idx] = pai;
  cHr[idx] = hr;  cHi[idx] = hi;
}

// Phase 2: serial combine across segments per channel -> per-segment prefix h.
__global__ __launch_bounds__(256) void k_scan_combine(
    const float* __restrict__ cAr, const float* __restrict__ cAi,
    const float* __restrict__ cHr, const float* __restrict__ cHi,
    float* __restrict__ hinR, float* __restrict__ hinI) {
  int ch = blockIdx.x * 256 + threadIdx.x;
  if (ch >= Bb * Dd) return;
  float hr = 0.f, hi = 0.f;
  for (int s = 0; s < SEG; s++) {
    int idx = s * (Bb * Dd) + ch;
    hinR[idx] = hr;
    hinI[idx] = hi;
    float ar = cAr[idx], ai = cAi[idx];
    float nhr = ar * hr - ai * hi + cHr[idx];
    float nhi = ar * hi + ai * hr + cHi[idx];
    hr = nhr; hi = nhi;
  }
}

// Phase 3: apply prefix, produce ycat = [q*h_re*silu(g) | q*h_im*silu(g)]
// packed along K (row stride 2*Dd) in bf16 for the fused complex projection.
__global__ __launch_bounds__(256) void k_scan_apply(
    const float4* __restrict__ qkvg, const float2* __restrict__ a2,
    const float* __restrict__ hinR, const float* __restrict__ hinI,
    unsigned short* __restrict__ ycat) {
  int tid  = blockIdx.x * 256 + threadIdx.x;
  int d    = tid & (Dd - 1);
  int rest = tid >> 10;
  int b    = rest & (Bb - 1);
  int s    = rest >> 1;
  if (s >= SEG) return;

  int ch  = b * Dd + d;
  int idx = s * (Bb * Dd) + ch;
  float hr = hinR[idx], hi = hinI[idx];
  size_t rowBase = (size_t)(b * Tt + s * SEGLEN);
  for (int t = 0; t < SEGLEN; t++) {
    size_t row = rowBase + t;
    float4 v4 = qkvg[row * Dd + d];     // q,k,v,g
    float2 av = a2[row * Dd + d];
    float acr, aci;
    ac_from(av.x, av.y, acr, aci);
    float kv  = v4.y * v4.z;
    float nhr = acr * hr - aci * hi + kv;
    float nhi = acr * hi + aci * hr;
    hr = nhr; hi = nhi;
    float gate = v4.w / (1.0f + expf(-v4.w));  // silu(g)
    float f = v4.x * gate;                      // q * silu(g)
    size_t o = row * (size_t)(2 * Dd) + d;
    ycat[o]      = f2bf(f * hr);
    ycat[o + Dd] = f2bf(f * hi);
  }
}

// ---------------------------------------------------------------------------
// LayerNorm over D=1024: out = LN(xin + res); optional f32 and bf16 outputs.
// ---------------------------------------------------------------------------
__global__ __launch_bounds__(256) void k_layernorm(
    const float* __restrict__ xin, const float* __restrict__ res,
    const float* __restrict__ scale, const float* __restrict__ bias,
    float* __restrict__ outf, unsigned short* __restrict__ outbf) {
  __shared__ float s1[256], s2[256];
  size_t row = blockIdx.x;
  const float* xr = xin + row * Dd;
  const float* rr = res + row * Dd;
  float v[4];
  float sum = 0.f, sq = 0.f;
#pragma unroll
  for (int j = 0; j < 4; j++) {
    int i = threadIdx.x + j * 256;
    float t = xr[i] + rr[i];
    v[j] = t;
    sum += t;
    sq  += t * t;
  }
  s1[threadIdx.x] = sum;
  s2[threadIdx.x] = sq;
  __syncthreads();
  for (int off = 128; off > 0; off >>= 1) {
    if (threadIdx.x < off) {
      s1[threadIdx.x] += s1[threadIdx.x + off];
      s2[threadIdx.x] += s2[threadIdx.x + off];
    }
    __syncthreads();
  }
  float mean = s1[0] * (1.0f / Dd);
  float var  = s2[0] * (1.0f / Dd) - mean * mean;
  float rstd = rsqrtf(var + 1e-6f);
#pragma unroll
  for (int j = 0; j < 4; j++) {
    int i = threadIdx.x + j * 256;
    float o = (v[j] - mean) * rstd * scale[i] + bias[i];
    if (outf)  outf[row * Dd + i]  = o;
    if (outbf) outbf[row * Dd + i] = f2bf(o);
  }
}

// ---------------------------------------------------------------------------
// Host launcher
// ---------------------------------------------------------------------------
extern "C" void kernel_launch(void* const* d_in, const int* in_sizes, int n_in,
                              void* d_out, int out_size, void* d_ws,
                              size_t ws_size, hipStream_t stream) {
  (void)in_sizes; (void)n_in; (void)out_size; (void)ws_size;

  const float* x       = (const float*)d_in[0];
  const float* W_qkvg  = (const float*)d_in[1];
  const float* W_a     = (const float*)d_in[2];
  const float* W_or    = (const float*)d_in[3];
  const float* W_oi    = (const float*)d_in[4];
  const float* ln1_s   = (const float*)d_in[5];
  const float* ln1_b   = (const float*)d_in[6];
  const float* W1      = (const float*)d_in[7];
  const float* b1      = (const float*)d_in[8];
  const float* W2      = (const float*)d_in[9];
  const float* b2      = (const float*)d_in[10];
  const float* ln2_s   = (const float*)d_in[11];
  const float* ln2_b   = (const float*)d_in[12];

  char* p = (char*)d_ws;
  auto alloc = [&](size_t bytes) -> void* {
    void* r = (void*)p;
    p += (bytes + 255) & ~(size_t)255;
    return r;
  };

  const size_t M = Mrows;
  __bf16* xbf      = (__bf16*)alloc(M * Dd * 2);
  __bf16* wqkvg_t  = (__bf16*)alloc((size_t)4 * Dd * Dd * 2);
  __bf16* wa_t     = (__bf16*)alloc((size_t)2 * Dd * Dd * 2);
  __bf16* wo_cat   = (__bf16*)alloc((size_t)Dd * 2 * Dd * 2);  // [1024][2048]
  __bf16* w1_t     = (__bf16*)alloc((size_t)Ff * Dd * 2);
  __bf16* w2_t     = (__bf16*)alloc((size_t)Dd * Ff * 2);
  float*  qkvg     = (float*)alloc(M * (size_t)(4 * Dd) * 4);
  float*  a2buf    = (float*)alloc(M * (size_t)(2 * Dd) * 4);
  float*  cAr      = (float*)alloc((size_t)SEG * Bb * Dd * 4);
  float*  cAi      = (float*)alloc((size_t)SEG * Bb * Dd * 4);
  float*  cHr      = (float*)alloc((size_t)SEG * Bb * Dd * 4);
  float*  cHi      = (float*)alloc((size_t)SEG * Bb * Dd * 4);
  float*  hinR     = (float*)alloc((size_t)SEG * Bb * Dd * 4);
  float*  hinI     = (float*)alloc((size_t)SEG * Bb * Dd * 4);
  unsigned short* ycat = (unsigned short*)alloc(M * (size_t)(2 * Dd) * 2);
  float*  attn     = (float*)alloc(M * Dd * 4);
  float*  y1       = (float*)alloc(M * Dd * 4);
  unsigned short* y1bf = (unsigned short*)alloc(M * Dd * 2);
  unsigned short* hbf  = (unsigned short*)alloc(M * (size_t)Ff * 2);
  float*  out2     = (float*)alloc(M * Dd * 4);

  // 1) x -> bf16
  {
    int n4 = (int)(M * Dd / 4);
    k_convert_bf16<<<n4 / 256, 256, 0, stream>>>((const float4*)x,
                                                 (ushort4*)xbf, n4);
  }
  // 2) weight transposes (f32 [K,N] -> bf16 [N,K]); Wo pair fused+negated.
  k_transpose_bf16<<<dim3(4 * Dd / 32, Dd / 32), 256, 0, stream>>>(
      W_qkvg, wqkvg_t, Dd, 4 * Dd, Dd, 0, 1.0f);
  k_transpose_bf16<<<dim3(2 * Dd / 32, Dd / 32), 256, 0, stream>>>(
      W_a, wa_t, Dd, 2 * Dd, Dd, 0, 1.0f);
  k_transpose_bf16<<<dim3(Dd / 32, Dd / 32), 256, 0, stream>>>(
      W_or, wo_cat, Dd, Dd, 2 * Dd, 0, 1.0f);
  k_transpose_bf16<<<dim3(Dd / 32, Dd / 32), 256, 0, stream>>>(
      W_oi, wo_cat, Dd, Dd, 2 * Dd, Dd, -1.0f);
  k_transpose_bf16<<<dim3(Ff / 32, Dd / 32), 256, 0, stream>>>(
      W1, w1_t, Dd, Ff, Dd, 0, 1.0f);
  k_transpose_bf16<<<dim3(Dd / 32, Ff / 32), 256, 0, stream>>>(
      W2, w2_t, Ff, Dd, Ff, 0, 1.0f);

  // 3) qkvg = x @ W_qkvg   [8192 x 4096]
  k_gemm_bf16_wmma<EPI_STORE><<<dim3(4 * Dd / 128, M / 256), 256, 0, stream>>>(
      xbf, wqkvg_t, qkvg, nullptr, nullptr, (int)M, 4 * Dd, Dd);
  // 4) a = x @ W_a         [8192 x 2048]
  k_gemm_bf16_wmma<EPI_STORE><<<dim3(2 * Dd / 128, M / 256), 256, 0, stream>>>(
      xbf, wa_t, a2buf, nullptr, nullptr, (int)M, 2 * Dd, Dd);

  // 5-7) chunked complex gate-loop scan
  {
    int total = Bb * Dd * SEG;  // 65536
    k_scan_partial<<<total / 256, 256, 0, stream>>>(
        (const float4*)qkvg, (const float2*)a2buf, cAr, cAi, cHr, cHi);
    k_scan_combine<<<(Bb * Dd) / 256, 256, 0, stream>>>(cAr, cAi, cHr, cHi,
                                                        hinR, hinI);
    k_scan_apply<<<total / 256, 256, 0, stream>>>(
        (const float4*)qkvg, (const float2*)a2buf, hinR, hinI, ycat);
  }

  // 8) attn = [y_re|y_im] @ [W_or; -W_oi]   (K = 2048, one fused GEMM)
  k_gemm_bf16_wmma<EPI_STORE><<<dim3(Dd / 128, M / 256), 256, 0, stream>>>(
      (const __bf16*)ycat, wo_cat, attn, nullptr, nullptr, (int)M, Dd, 2 * Dd);

  // 9) y1 = LN(attn + x)
  k_layernorm<<<(int)M, 256, 0, stream>>>(attn, x, ln1_s, ln1_b, y1, y1bf);

  // 10) h = gelu(y1 @ W1 + b1) -> bf16   [8192 x 4096]
  k_gemm_bf16_wmma<EPI_BIAS_GELU_BF16>
      <<<dim3(Ff / 128, M / 256), 256, 0, stream>>>(
          (const __bf16*)y1bf, w1_t, nullptr, hbf, b1, (int)M, Ff, Dd);

  // 11) out2 = h @ W2 + b2               [8192 x 1024], K=4096
  k_gemm_bf16_wmma<EPI_BIAS_F32><<<dim3(Dd / 128, M / 256), 256, 0, stream>>>(
      (const __bf16*)hbf, w2_t, out2, nullptr, b2, (int)M, Dd, Ff);

  // 12) out = LN(out2 + y1)
  k_layernorm<<<(int)M, 256, 0, stream>>>(out2, y1, ln2_s, ln2_b,
                                          (float*)d_out, nullptr);
}
